// Attn_user_90795608637906
// MI455X (gfx1250) — compile-verified, matrix operand order checked by reading, not run on hardware
//
#include <hip/hip_runtime.h>
#include <hip/hip_bf16.h>

typedef __bf16 bf16;
typedef __attribute__((ext_vector_type(16))) __bf16 v16bf;
typedef __attribute__((ext_vector_type(8)))  __bf16 v8bf;
typedef __attribute__((ext_vector_type(8)))  float   v8f;
typedef __attribute__((ext_vector_type(4)))  float   v4f;
typedef __attribute__((ext_vector_type(4)))  int     v4i;

#define M_DIM 8192
#define N_DIM 8192
#define K_DIM 1024

#if __has_builtin(__builtin_amdgcn_global_load_async_to_lds_b128)
#define HAS_ASYNC_LDS 1
#else
#define HAS_ASYNC_LDS 0
#endif

// 16-byte global->LDS copy: async (ASYNCcnt) when available, sync fallback.
// Probe-learned signature: (v4i* gsrc, v4i* ldsdst, int imm_offset, int cpol).
__device__ __forceinline__ void cp16(const bf16* g, bf16* l) {
#if HAS_ASYNC_LDS
    __builtin_amdgcn_global_load_async_to_lds_b128((v4i*)g, (v4i*)l, 0, 0);
#else
    *(v8bf*)l = *(const v8bf*)g;
#endif
}

__device__ __forceinline__ void wait_async0() {
#if HAS_ASYNC_LDS
#if __has_builtin(__builtin_amdgcn_s_wait_asynccnt)
    __builtin_amdgcn_s_wait_asynccnt(0);
#else
    asm volatile("s_wait_asynccnt 0x0" ::: "memory");
#endif
#endif
}

// ---------------------------------------------------------------------------
// f32 -> bf16 conversion pass (one sweep; result lives in L2-resident d_ws)
// ---------------------------------------------------------------------------
__global__ __launch_bounds__(256) void cvt_f32_to_bf16(const float* __restrict__ in,
                                                       bf16* __restrict__ out,
                                                       int n_vec8) {
    int v = blockIdx.x * 256 + threadIdx.x;
    if (v >= n_vec8) return;
    int i = v * 8;
    v4f a = *(const v4f*)(in + i);
    v4f b = *(const v4f*)(in + i + 4);
    v8bf o;
#pragma unroll
    for (int k = 0; k < 4; ++k) {
        o[k]     = (bf16)a[k];
        o[k + 4] = (bf16)b[k];
    }
    *(v8bf*)(out + i) = o;
}

// ---------------------------------------------------------------------------
// Fragment builders, ISA 7.12.2 16-bit 16x32 layout:
//   lane<16 : row = base+lane,    K = {k0..k0+7, k0+16..k0+23}
//   lane>=16: row = base+lane-16, K = {k0+8..k0+15, k0+24..k0+31}
// ---------------------------------------------------------------------------
#define LDS_STRIDE 40   // 32 data elems + 8 pad; 80B rows, 16B-aligned segments

__device__ __forceinline__ v16bf frag_from_lds(const bf16* tile, int row0, int lane) {
    const bf16* p = tile + (row0 + (lane & 15)) * LDS_STRIDE + ((lane & 16) ? 8 : 0);
    v8bf lo = *(const v8bf*)(p);
    v8bf hi = *(const v8bf*)(p + 16);
    v16bf f;
#pragma unroll
    for (int i = 0; i < 8; ++i) { f[i] = lo[i]; f[i + 8] = hi[i]; }
    return f;
}

__device__ __forceinline__ v16bf load_frag_f32(const float* __restrict__ base,
                                               int row0, int k0, int lane) {
    int r  = row0 + (lane & 15);
    int kk = k0 + ((lane & 16) ? 8 : 0);
    const float* p = base + (size_t)r * K_DIM + kk;
    v4f a0 = *(const v4f*)(p);
    v4f a1 = *(const v4f*)(p + 4);
    v4f a2 = *(const v4f*)(p + 16);
    v4f a3 = *(const v4f*)(p + 20);
    v16bf f;
#pragma unroll
    for (int i = 0; i < 4; ++i) {
        f[i]      = (bf16)a0[i];
        f[i + 4]  = (bf16)a1[i];
        f[i + 8]  = (bf16)a2[i];
        f[i + 12] = (bf16)a3[i];
    }
    return f;
}

// ---------------------------------------------------------------------------
// GEMM: block = 256 threads = 8 waves (2 M-waves x 4 N-waves).
// Block tile 64(M) x 256(N); wave tile 32x64 = 2x4 accumulators.
// PRE path: double-buffered LDS staging fed by async global->LDS copies;
// per K-step of 32: 5 async 16B copies/thread, 12 ds_load_b128 + 8 WMMA /wave.
// ---------------------------------------------------------------------------
template <bool PRE>
__global__ __launch_bounds__(256, 2) void gemm_bf16_nt(const void* __restrict__ Av,
                                                       const void* __restrict__ Bv,
                                                       float* __restrict__ out) {
    __shared__ bf16 sA[2][64  * LDS_STRIDE];   // 2 x 5 KB
    __shared__ bf16 sB[2][256 * LDS_STRIDE];   // 2 x 20 KB

    const int lane  = threadIdx.x & 31;
    const int wave  = threadIdx.x >> 5;
    const int waveM = wave & 1;
    const int waveN = wave >> 1;
    const int m0 = blockIdx.y * 64 + waveM * 32;
    const int n0 = blockIdx.x * 256 + waveN * 64;

    v8f acc[2][4];
#pragma unroll
    for (int i = 0; i < 2; ++i)
#pragma unroll
        for (int j = 0; j < 4; ++j) acc[i][j] = (v8f){0.f, 0.f, 0.f, 0.f, 0.f, 0.f, 0.f, 0.f};

    if constexpr (PRE) {
        const bf16* Ab = (const bf16*)Av;
        const bf16* Bb = (const bf16*)Bv;

        // Per-thread copy slot: 256 threads cover 64 rows x 4 segments of 16B.
        const int t   = threadIdx.x;
        const int seg = (t & 3) * 8;   // element offset within 32-elem K slab
        const int rr  = t >> 2;        // 0..63
        const bf16* gA = Ab + (size_t)(blockIdx.y * 64  + rr) * K_DIM + seg;
        const bf16* gB = Bb + (size_t)(blockIdx.x * 256 + rr) * K_DIM + seg;
        bf16* lA[2] = { &sA[0][rr * LDS_STRIDE + seg], &sA[1][rr * LDS_STRIDE + seg] };
        bf16* lB[2] = { &sB[0][rr * LDS_STRIDE + seg], &sB[1][rr * LDS_STRIDE + seg] };

        auto issue = [&](int buf, int k0) {
            cp16(gA + k0, lA[buf]);                       // A: 64 rows
#pragma unroll
            for (int i = 0; i < 4; ++i)                   // B: 4 x 64 rows
                cp16(gB + k0 + (size_t)(i * 64) * K_DIM, lB[buf] + i * 64 * LDS_STRIDE);
        };

        issue(0, 0);
        constexpr int NSTEP = K_DIM / 32;
#pragma unroll 2
        for (int ks = 0; ks < NSTEP; ++ks) {
            wait_async0();        // our stage-ks copies have landed
            __syncthreads();      // everyone's copies visible; prev-buf readers done
            if (ks + 1 < NSTEP) issue((ks + 1) & 1, (ks + 1) * 32);

            const bf16* tA = sA[ks & 1];
            const bf16* tB = sB[ks & 1];
            v16bf a[2], b[4];
#pragma unroll
            for (int i = 0; i < 2; ++i) a[i] = frag_from_lds(tA, waveM * 32 + 16 * i, lane);
#pragma unroll
            for (int j = 0; j < 4; ++j) b[j] = frag_from_lds(tB, waveN * 64 + 16 * j, lane);
#pragma unroll
            for (int i = 0; i < 2; ++i)
#pragma unroll
                for (int j = 0; j < 4; ++j)
                    acc[i][j] = __builtin_amdgcn_wmma_f32_16x16x32_bf16(
                        false, a[i], false, b[j], (short)0, acc[i][j], false, false);
        }
    } else {
        const float* Af = (const float*)Av;
        const float* Bf = (const float*)Bv;
#pragma unroll 2
        for (int k0 = 0; k0 < K_DIM; k0 += 32) {
            v16bf a[2], b[4];
#pragma unroll
            for (int i = 0; i < 2; ++i) a[i] = load_frag_f32(Af, m0 + 16 * i, k0, lane);
#pragma unroll
            for (int j = 0; j < 4; ++j) b[j] = load_frag_f32(Bf, n0 + 16 * j, k0, lane);
            if (k0 + 32 < K_DIM) {
                __builtin_prefetch(Af + (size_t)(m0 + (lane & 15)) * K_DIM + k0 + 32, 0, 3);
                __builtin_prefetch(Bf + (size_t)(n0 + (lane & 15)) * K_DIM + k0 + 32, 0, 3);
            }
#pragma unroll
            for (int i = 0; i < 2; ++i)
#pragma unroll
                for (int j = 0; j < 4; ++j)
                    acc[i][j] = __builtin_amdgcn_wmma_f32_16x16x32_bf16(
                        false, a[i], false, b[j], (short)0, acc[i][j], false, false);
        }
    }

    // C/D layout: VGPR r holds M=r (lanes 0-15) and M=r+8 (lanes 16-31); N = lane&15.
    const int rowAdd = (lane & 16) ? 8 : 0;
    const int col0   = lane & 15;
#pragma unroll
    for (int i = 0; i < 2; ++i)
#pragma unroll
        for (int j = 0; j < 4; ++j) {
            const int col = n0 + 16 * j + col0;
#pragma unroll
            for (int r = 0; r < 8; ++r) {
                const int row = m0 + 16 * i + r + rowAdd;
                out[(size_t)row * N_DIM + col] = acc[i][j][r];
            }
        }
}

// ---------------------------------------------------------------------------
// Row softmax: one block per row; 32KB row cache in LDS (WGP has 320KB).
// ---------------------------------------------------------------------------
#define SM_T 256
__global__ __launch_bounds__(SM_T) void softmax_rows(float* __restrict__ data) {
    __shared__ float row[N_DIM];
    __shared__ float red[SM_T / 32];
    __shared__ float bval;

    const int lane = threadIdx.x & 31;
    const int wave = threadIdx.x >> 5;
    float* g = data + (size_t)blockIdx.x * N_DIM;

    float lmax = -3.402823466e38f;
    for (int i = threadIdx.x; i < N_DIM; i += SM_T) {
        float v = g[i];
        row[i] = v;
        lmax = fmaxf(lmax, v);
    }
#pragma unroll
    for (int off = 16; off; off >>= 1) lmax = fmaxf(lmax, __shfl_xor(lmax, off, 32));
    if (lane == 0) red[wave] = lmax;
    __syncthreads();
    if (threadIdx.x == 0) {
        float m = red[0];
#pragma unroll
        for (int w = 1; w < SM_T / 32; ++w) m = fmaxf(m, red[w]);
        bval = m;
    }
    __syncthreads();
    const float m = bval;

    float lsum = 0.f;
    for (int i = threadIdx.x; i < N_DIM; i += SM_T) {
        float e = __expf(row[i] - m);
        row[i] = e;
        lsum += e;
    }
#pragma unroll
    for (int off = 16; off; off >>= 1) lsum += __shfl_xor(lsum, off, 32);
    if (lane == 0) red[wave] = lsum;
    __syncthreads();
    if (threadIdx.x == 0) {
        float s = red[0];
#pragma unroll
        for (int w = 1; w < SM_T / 32; ++w) s += red[w];
        bval = 1.f / s;
    }
    __syncthreads();
    const float inv = bval;

    for (int i = threadIdx.x; i < N_DIM; i += SM_T) g[i] = row[i] * inv;
}

// ---------------------------------------------------------------------------
extern "C" void kernel_launch(void* const* d_in, const int* in_sizes, int n_in,
                              void* d_out, int out_size, void* d_ws, size_t ws_size,
                              hipStream_t stream) {
    const float* A = (const float*)d_in[0];  // user_emb [8192,1024]
    const float* B = (const float*)d_in[1];  // id_emb   [8192,1024]
    float* out = (float*)d_out;              // [8192,8192]

    const size_t elems = (size_t)M_DIM * K_DIM;            // 8M per matrix
    const bool pre = ws_size >= 2 * elems * sizeof(bf16);  // 32 MB scratch needed

    dim3 grid(N_DIM / 256, M_DIM / 64);
    dim3 block(256);

    if (pre) {
        bf16* Abf = (bf16*)d_ws;
        bf16* Bbf = Abf + elems;
        const int nv = (int)(elems / 8);
        cvt_f32_to_bf16<<<(nv + 255) / 256, 256, 0, stream>>>(A, Abf, nv);
        cvt_f32_to_bf16<<<(nv + 255) / 256, 256, 0, stream>>>(B, Bbf, nv);
        gemm_bf16_nt<true><<<grid, block, 0, stream>>>(Abf, Bbf, out);
    } else {
        gemm_bf16_nt<false><<<grid, block, 0, stream>>>(A, B, out);
    }
    softmax_rows<<<M_DIM, SM_T, 0, stream>>>(out);
}
